// TransformerDecoderLayer_77343771066480
// MI455X (gfx1250) — compile-verified, matrix operand order checked by reading, not run on hardware
//
#include <hip/hip_runtime.h>
#include <cmath>

typedef __bf16 bf16_t;
typedef __attribute__((ext_vector_type(16))) __bf16 v16bf;
typedef __attribute__((ext_vector_type(8)))  float  v8f;

#define D_MODEL 1024
#define NHEAD   16
#define DHEAD   64
#define SEQ     2048
#define BATCH   2
#define MTOK    (BATCH*SEQ)
#define NEGBIG  (-1.0e9f)
#define LN_EPS  1e-5f

static __device__ __forceinline__ v8f zero8() {
    v8f z;
#pragma unroll
    for (int i = 0; i < 8; ++i) z[i] = 0.f;
    return z;
}

// async 16B global -> LDS copy (CDNA5, tracked by ASYNCcnt)
static __device__ __forceinline__ void async_cp16(unsigned lds_off, const bf16_t* g) {
    asm volatile("global_load_async_to_lds_b128 %0, %1, off" :: "v"(lds_off), "v"(g) : "memory");
}
static __device__ __forceinline__ void wait_async0() {
    asm volatile("s_wait_asynccnt 0" ::: "memory");
}
static __device__ __forceinline__ unsigned lds_off32(const void* p) {
    return (unsigned)(uintptr_t)p;   // flat LDS address keeps offset in addr[31:0]
}

// ---------------------------------------------------------------- fp32 -> bf16
__global__ void cvt_f32_bf16(const float* __restrict__ src, bf16_t* __restrict__ dst, int n) {
    int i = blockIdx.x * blockDim.x + threadIdx.x;
    if (i < n) dst[i] = (bf16_t)src[i];
}

// ------------------------------------------------- embedding + LayerNorm (ln1)
__global__ __launch_bounds__(256) void embed_ln_k(
    const int* __restrict__ ids, const float* __restrict__ tok_emb,
    const float* __restrict__ pos_emb, const float* __restrict__ gamma,
    const float* __restrict__ beta, float* __restrict__ outF, bf16_t* __restrict__ outB)
{
    __shared__ float s1[256], s2[256];
    const int tok = blockIdx.x;
    const int srow = tok % SEQ;
    const int id = ids[tok];
    const float* te = tok_emb + (size_t)id * D_MODEL;
    const float* pe = pos_emb + (size_t)srow * D_MODEL;
    float v[4]; float lsum = 0.f, lsq = 0.f;
#pragma unroll
    for (int i = 0; i < 4; ++i) {
        int c = threadIdx.x + 256 * i;
        float x = te[c] + pe[c];
        v[i] = x; lsum += x; lsq += x * x;
    }
    s1[threadIdx.x] = lsum; s2[threadIdx.x] = lsq; __syncthreads();
    for (int off = 128; off > 0; off >>= 1) {
        if (threadIdx.x < off) { s1[threadIdx.x] += s1[threadIdx.x + off]; s2[threadIdx.x] += s2[threadIdx.x + off]; }
        __syncthreads();
    }
    float mu = s1[0] * (1.0f / D_MODEL);
    float var = s2[0] * (1.0f / D_MODEL) - mu * mu;
    float rstd = rsqrtf(var + LN_EPS);
    float* of = outF + (size_t)tok * D_MODEL;
    bf16_t* ob = outB + (size_t)tok * D_MODEL;
#pragma unroll
    for (int i = 0; i < 4; ++i) {
        int c = threadIdx.x + 256 * i;
        float y = (v[i] - mu) * rstd * gamma[c] + beta[c];
        of[c] = y; ob[c] = (bf16_t)y;
    }
}

// ------------------------------------------------------------------- LayerNorm
__global__ __launch_bounds__(256) void layernorm_k(
    const float* __restrict__ in, const float* __restrict__ gamma,
    const float* __restrict__ beta, bf16_t* __restrict__ outB)
{
    __shared__ float s1[256], s2[256];
    const int tok = blockIdx.x;
    const float* row = in + (size_t)tok * D_MODEL;
    float v[4]; float lsum = 0.f, lsq = 0.f;
#pragma unroll
    for (int i = 0; i < 4; ++i) {
        int c = threadIdx.x + 256 * i;
        float x = row[c]; v[i] = x; lsum += x; lsq += x * x;
    }
    s1[threadIdx.x] = lsum; s2[threadIdx.x] = lsq; __syncthreads();
    for (int off = 128; off > 0; off >>= 1) {
        if (threadIdx.x < off) { s1[threadIdx.x] += s1[threadIdx.x + off]; s2[threadIdx.x] += s2[threadIdx.x + off]; }
        __syncthreads();
    }
    float mu = s1[0] * (1.0f / D_MODEL);
    float var = s2[0] * (1.0f / D_MODEL) - mu * mu;
    float rstd = rsqrtf(var + LN_EPS);
    bf16_t* ob = outB + (size_t)tok * D_MODEL;
#pragma unroll
    for (int i = 0; i < 4; ++i) {
        int c = threadIdx.x + 256 * i;
        ob[c] = (bf16_t)((v[i] - mu) * rstd * gamma[c] + beta[c]);
    }
}

// -------------------------------------------------------------- bf16 WMMA GEMM
// C[M,N] = A[M,K]*W[K,N] (+bias)(+GELU)(+res); f32 and/or bf16 out (opt. bf16^T).
// Block 256 thr (8 waves); block tile 128x128, wave tile 32x64, BK=32,
// double-buffered LDS; A staged via global_load_async_to_lds_b128.
__global__ __launch_bounds__(256) void gemm_bf16_k(
    const bf16_t* __restrict__ A, const bf16_t* __restrict__ W,
    const float* __restrict__ bias, const float* __restrict__ res,
    float* __restrict__ outF, bf16_t* __restrict__ outB,
    int M, int N, int K, int applyGelu, int outBTrans)
{
    __shared__ __align__(16) bf16_t As[2][128][32];   // [buf][m][k]
    __shared__ __align__(16) bf16_t Bs[2][128][32];   // [buf][n][k] (transposed)
    const int tid = threadIdx.x;
    const int lane = tid & 31, wave = tid >> 5;
    const int half = lane >> 4, mr = lane & 15;
    const int wr = wave >> 1, wc = wave & 1;
    const int m0 = blockIdx.y * 128, n0 = blockIdx.x * 128;

    // A staging: 2x16B per thread
    const int arow = tid >> 2, acol = (tid & 3) * 8;
    const bf16_t* ag0 = A + (size_t)(m0 + arow) * K + acol;
    const bf16_t* ag1 = A + (size_t)(m0 + 64 + arow) * K + acol;
    const unsigned la0b0 = lds_off32(&As[0][arow][acol]);
    const unsigned la0b1 = lds_off32(&As[1][arow][acol]);
    const unsigned la1b0 = lds_off32(&As[0][64 + arow][acol]);
    const unsigned la1b1 = lds_off32(&As[1][64 + arow][acol]);
    // B staging: 2x16B per thread, regs then transposed LDS store
    const int brow = tid >> 4, bcol = (tid & 15) * 8;
    const bf16_t* bg0 = W + (size_t)brow * N + n0 + bcol;
    const bf16_t* bg1 = W + (size_t)(16 + brow) * N + n0 + bcol;

    v8f acc[2][4];
#pragma unroll
    for (int i = 0; i < 2; ++i)
#pragma unroll
        for (int j = 0; j < 4; ++j) acc[i][j] = zero8();

    const int nk = K >> 5;
    // prologue: stage k-step 0 into buffer 0
    async_cp16(la0b0, ag0);
    async_cp16(la1b0, ag1);
    ag0 += 32; ag1 += 32;
    uint4 br0 = *(const uint4*)bg0;
    uint4 br1 = *(const uint4*)bg1;
    bg0 += (size_t)32 * N; bg1 += (size_t)32 * N;

    for (int it = 0; it < nk; ++it) {
        const int cur = it & 1;
        wait_async0();                    // own async A copies for buf[cur] done
        {                                 // B regs -> LDS transposed
            bf16_t t[8];
            *(uint4*)t = br0;
#pragma unroll
            for (int j = 0; j < 8; ++j) Bs[cur][bcol + j][brow] = t[j];
            *(uint4*)t = br1;
#pragma unroll
            for (int j = 0; j < 8; ++j) Bs[cur][bcol + j][16 + brow] = t[j];
        }
        __syncthreads();                  // buf[cur] fully populated for all waves

        if (it + 1 < nk) {                // prefetch next tile into buf[1-cur]
            async_cp16(cur ? la0b0 : la0b1, ag0);
            async_cp16(cur ? la1b0 : la1b1, ag1);
            ag0 += 32; ag1 += 32;
            br0 = *(const uint4*)bg0;
            br1 = *(const uint4*)bg1;
            bg0 += (size_t)32 * N; bg1 += (size_t)32 * N;
        }

        // preload all fragments into distinct registers so the compiler can
        // issue every ds_load_b128 up front and use partial s_wait_dscnt waits
        v16bf af0, af1, bfr0, bfr1, bfr2, bfr3;
        ((uint4*)&af0)[0]  = *(const uint4*)(&As[cur][wr * 32 + mr][half * 8]);
        ((uint4*)&af0)[1]  = *(const uint4*)(&As[cur][wr * 32 + mr][16 + half * 8]);
        ((uint4*)&af1)[0]  = *(const uint4*)(&As[cur][wr * 32 + 16 + mr][half * 8]);
        ((uint4*)&af1)[1]  = *(const uint4*)(&As[cur][wr * 32 + 16 + mr][16 + half * 8]);
        ((uint4*)&bfr0)[0] = *(const uint4*)(&Bs[cur][wc * 64 + 0 * 16 + mr][half * 8]);
        ((uint4*)&bfr0)[1] = *(const uint4*)(&Bs[cur][wc * 64 + 0 * 16 + mr][16 + half * 8]);
        ((uint4*)&bfr1)[0] = *(const uint4*)(&Bs[cur][wc * 64 + 1 * 16 + mr][half * 8]);
        ((uint4*)&bfr1)[1] = *(const uint4*)(&Bs[cur][wc * 64 + 1 * 16 + mr][16 + half * 8]);
        ((uint4*)&bfr2)[0] = *(const uint4*)(&Bs[cur][wc * 64 + 2 * 16 + mr][half * 8]);
        ((uint4*)&bfr2)[1] = *(const uint4*)(&Bs[cur][wc * 64 + 2 * 16 + mr][16 + half * 8]);
        ((uint4*)&bfr3)[0] = *(const uint4*)(&Bs[cur][wc * 64 + 3 * 16 + mr][half * 8]);
        ((uint4*)&bfr3)[1] = *(const uint4*)(&Bs[cur][wc * 64 + 3 * 16 + mr][16 + half * 8]);

        acc[0][0] = __builtin_amdgcn_wmma_f32_16x16x32_bf16(false, af0, false, bfr0, (short)0, acc[0][0], false, false);
        acc[1][0] = __builtin_amdgcn_wmma_f32_16x16x32_bf16(false, af1, false, bfr0, (short)0, acc[1][0], false, false);
        acc[0][1] = __builtin_amdgcn_wmma_f32_16x16x32_bf16(false, af0, false, bfr1, (short)0, acc[0][1], false, false);
        acc[1][1] = __builtin_amdgcn_wmma_f32_16x16x32_bf16(false, af1, false, bfr1, (short)0, acc[1][1], false, false);
        acc[0][2] = __builtin_amdgcn_wmma_f32_16x16x32_bf16(false, af0, false, bfr2, (short)0, acc[0][2], false, false);
        acc[1][2] = __builtin_amdgcn_wmma_f32_16x16x32_bf16(false, af1, false, bfr2, (short)0, acc[1][2], false, false);
        acc[0][3] = __builtin_amdgcn_wmma_f32_16x16x32_bf16(false, af0, false, bfr3, (short)0, acc[0][3], false, false);
        acc[1][3] = __builtin_amdgcn_wmma_f32_16x16x32_bf16(false, af1, false, bfr3, (short)0, acc[1][3], false, false);
        // no trailing barrier: buffers alternate; next write to buf[cur] is
        // after the next iteration's collective barrier.
    }

#pragma unroll
    for (int tm = 0; tm < 2; ++tm)
#pragma unroll
        for (int tn = 0; tn < 4; ++tn)
#pragma unroll
            for (int v = 0; v < 8; ++v) {
                const int m = m0 + wr * 32 + tm * 16 + v + 8 * half;
                const int n = n0 + wc * 64 + tn * 16 + mr;
                float val = acc[tm][tn][v];
                if (bias) val += bias[n];
                if (applyGelu) val = 0.5f * val * (1.0f + erff(val * 0.70710678118f));
                if (res) val += res[(size_t)m * N + n];
                if (outF) outF[(size_t)m * N + n] = val;
                if (outB) {
                    if (outBTrans) outB[(size_t)n * M + m] = (bf16_t)val;
                    else           outB[(size_t)m * N + n] = (bf16_t)val;
                }
            }
}

// ------------------------------------------------------ flash attention (WMMA)
// One wave per 16-row query tile of one (b,h). Qp/Kp: [tok][h*64+d] bf16,
// VpT: [h*64+d][tok] bf16 (transposed). outB: [tok][h*64+d] bf16.
__global__ __launch_bounds__(256) void flash_attn_k(
    const bf16_t* __restrict__ Qp, const bf16_t* __restrict__ Kp,
    const bf16_t* __restrict__ VpT, const int* __restrict__ ids,
    bf16_t* __restrict__ outB, int causal)
{
    __shared__ __align__(16) bf16_t Pbuf[8][16][32];
    const int tid = threadIdx.x, w = tid >> 5, lane = tid & 31;
    const int half = lane >> 4, mr = lane & 15;
    const int nQT = SEQ / 16;
    const int g = blockIdx.x * 8 + w;
    const int qt = g % nQT;
    const int h  = (g / nQT) % NHEAD;
    const int b  = g / (nQT * NHEAD);
    const int q0 = qt * 16;

    v16bf aq0, aq1;
    {
        const bf16_t* qrow = Qp + (size_t)(b * SEQ + q0 + mr) * D_MODEL + h * DHEAD;
        ((uint4*)&aq0)[0] = *(const uint4*)(qrow + half * 8);
        ((uint4*)&aq0)[1] = *(const uint4*)(qrow + 16 + half * 8);
        ((uint4*)&aq1)[0] = *(const uint4*)(qrow + 32 + half * 8);
        ((uint4*)&aq1)[1] = *(const uint4*)(qrow + 48 + half * 8);
    }

    v8f o0 = zero8(), o1 = zero8(), o2 = zero8(), o3 = zero8();
    float l[8], mx[8];
#pragma unroll
    for (int v = 0; v < 8; ++v) { l[v] = 0.f; mx[v] = -3.0e38f; }

    const int kend = causal ? (q0 + 16) : SEQ;
    for (int kb = 0; kb < kend; kb += 32) {
        v8f s0 = zero8(), s1 = zero8();
        {
            const bf16_t* kr0 = Kp + (size_t)(b * SEQ + kb + mr) * D_MODEL + h * DHEAD;
            const bf16_t* kr1 = kr0 + (size_t)16 * D_MODEL;
            v16bf bk;
            ((uint4*)&bk)[0] = *(const uint4*)(kr0 + half * 8);
            ((uint4*)&bk)[1] = *(const uint4*)(kr0 + 16 + half * 8);
            s0 = __builtin_amdgcn_wmma_f32_16x16x32_bf16(false, aq0, false, bk, (short)0, s0, false, false);
            ((uint4*)&bk)[0] = *(const uint4*)(kr0 + 32 + half * 8);
            ((uint4*)&bk)[1] = *(const uint4*)(kr0 + 48 + half * 8);
            s0 = __builtin_amdgcn_wmma_f32_16x16x32_bf16(false, aq1, false, bk, (short)0, s0, false, false);
            ((uint4*)&bk)[0] = *(const uint4*)(kr1 + half * 8);
            ((uint4*)&bk)[1] = *(const uint4*)(kr1 + 16 + half * 8);
            s1 = __builtin_amdgcn_wmma_f32_16x16x32_bf16(false, aq0, false, bk, (short)0, s1, false, false);
            ((uint4*)&bk)[0] = *(const uint4*)(kr1 + 32 + half * 8);
            ((uint4*)&bk)[1] = *(const uint4*)(kr1 + 48 + half * 8);
            s1 = __builtin_amdgcn_wmma_f32_16x16x32_bf16(false, aq1, false, bk, (short)0, s1, false, false);
        }
        const int key0 = kb + mr, key1 = kb + 16 + mr;
        const float pm0 = (ids[b * SEQ + key0] == 0) ? NEGBIG : 0.f;
        const float pm1 = (ids[b * SEQ + key1] == 0) ? NEGBIG : 0.f;

#pragma unroll
        for (int v = 0; v < 8; ++v) {
            const int qg = q0 + v + 8 * half;
            float e0 = s0[v] * 0.125f + pm0 + ((causal && key0 > qg) ? NEGBIG : 0.f);
            float e1 = s1[v] * 0.125f + pm1 + ((causal && key1 > qg) ? NEGBIG : 0.f);
            float rm = fmaxf(e0, e1);
#pragma unroll
            for (int off = 1; off < 16; off <<= 1) rm = fmaxf(rm, __shfl_xor(rm, off, 32));
            float mnew = fmaxf(mx[v], rm);
            float corr = __expf(mx[v] - mnew);
            mx[v] = mnew;
            float p0 = __expf(e0 - mnew), p1 = __expf(e1 - mnew);
            float rs = p0 + p1;
#pragma unroll
            for (int off = 1; off < 16; off <<= 1) rs += __shfl_xor(rs, off, 32);
            l[v] = l[v] * corr + rs;
            o0[v] *= corr; o1[v] *= corr; o2[v] *= corr; o3[v] *= corr;
            Pbuf[w][v + 8 * half][mr]      = (bf16_t)p0;
            Pbuf[w][v + 8 * half][mr + 16] = (bf16_t)p1;
        }
        // wave-local LDS transpose: make stores visible before fragment loads
        asm volatile("s_wait_dscnt 0" ::: "memory");

        v16bf ap;
        ((uint4*)&ap)[0] = *(const uint4*)(&Pbuf[w][mr][half * 8]);
        ((uint4*)&ap)[1] = *(const uint4*)(&Pbuf[w][mr][16 + half * 8]);

        const bf16_t* vbase = VpT + (size_t)(h * DHEAD) * MTOK + (size_t)b * SEQ + kb;
        v16bf bv;
        {
            const bf16_t* vr = vbase + (size_t)(0 * 16 + mr) * MTOK;
            ((uint4*)&bv)[0] = *(const uint4*)(vr + half * 8);
            ((uint4*)&bv)[1] = *(const uint4*)(vr + 16 + half * 8);
            o0 = __builtin_amdgcn_wmma_f32_16x16x32_bf16(false, ap, false, bv, (short)0, o0, false, false);
        }
        {
            const bf16_t* vr = vbase + (size_t)(1 * 16 + mr) * MTOK;
            ((uint4*)&bv)[0] = *(const uint4*)(vr + half * 8);
            ((uint4*)&bv)[1] = *(const uint4*)(vr + 16 + half * 8);
            o1 = __builtin_amdgcn_wmma_f32_16x16x32_bf16(false, ap, false, bv, (short)0, o1, false, false);
        }
        {
            const bf16_t* vr = vbase + (size_t)(2 * 16 + mr) * MTOK;
            ((uint4*)&bv)[0] = *(const uint4*)(vr + half * 8);
            ((uint4*)&bv)[1] = *(const uint4*)(vr + 16 + half * 8);
            o2 = __builtin_amdgcn_wmma_f32_16x16x32_bf16(false, ap, false, bv, (short)0, o2, false, false);
        }
        {
            const bf16_t* vr = vbase + (size_t)(3 * 16 + mr) * MTOK;
            ((uint4*)&bv)[0] = *(const uint4*)(vr + half * 8);
            ((uint4*)&bv)[1] = *(const uint4*)(vr + 16 + half * 8);
            o3 = __builtin_amdgcn_wmma_f32_16x16x32_bf16(false, ap, false, bv, (short)0, o3, false, false);
        }
    }

#pragma unroll
    for (int v = 0; v < 8; ++v) {
        float inv = 1.0f / l[v];
        int tokrow = b * SEQ + q0 + v + 8 * half;
        bf16_t* orow = outB + (size_t)tokrow * D_MODEL + h * DHEAD + mr;
        orow[0]  = (bf16_t)(o0[v] * inv);
        orow[16] = (bf16_t)(o1[v] * inv);
        orow[32] = (bf16_t)(o2[v] * inv);
        orow[48] = (bf16_t)(o3[v] * inv);
    }
}

// ----------------------------------------------------------------------- host
extern "C" void kernel_launch(void* const* d_in, const int* in_sizes, int n_in,
                              void* d_out, int out_size, void* d_ws, size_t ws_size,
                              hipStream_t stream)
{
    (void)in_sizes; (void)n_in; (void)out_size; (void)ws_size;
    const float* input_embedding = (const float*)d_in[0];
    const int*   input_ids       = (const int*)  d_in[1];
    const int*   target_ids      = (const int*)  d_in[2];
    const float* tok_emb         = (const float*)d_in[3];
    const float* pos_emb         = (const float*)d_in[4];
    const float* ln1_g = (const float*)d_in[5],  *ln1_b = (const float*)d_in[6];
    const float* q1_w  = (const float*)d_in[7],  *q1_b  = (const float*)d_in[8];
    const float* k1_w  = (const float*)d_in[9],  *k1_b  = (const float*)d_in[10];
    const float* v1_w  = (const float*)d_in[11], *v1_b  = (const float*)d_in[12];
    const float* out1_w= (const float*)d_in[13], *out1_b= (const float*)d_in[14];
    const float* ln2_g = (const float*)d_in[15], *ln2_b = (const float*)d_in[16];
    const float* q2_w  = (const float*)d_in[17], *q2_b  = (const float*)d_in[18];
    const float* k2_w  = (const float*)d_in[19], *k2_b  = (const float*)d_in[20];
    const float* v2_w  = (const float*)d_in[21], *v2_b  = (const float*)d_in[22];
    const float* out2_w= (const float*)d_in[23], *out2_b= (const float*)d_in[24];
    const float* ln3_g = (const float*)d_in[25], *ln3_b = (const float*)d_in[26];
    const float* mlp_w1= (const float*)d_in[27], *mlp_b1= (const float*)d_in[28];
    const float* mlp_w2= (const float*)d_in[29], *mlp_b2= (const float*)d_in[30];

    char* p = (char*)d_ws;
    auto alloc = [&](size_t bytes) -> char* {
        char* r = p; p += (bytes + 255) & ~(size_t)255; return r;
    };
    const size_t W1 = (size_t)D_MODEL * D_MODEL;      // 1M
    const size_t WM = (size_t)D_MODEL * 4 * D_MODEL;  // 4M
    bf16_t* wq1  = (bf16_t*)alloc(W1 * 2);
    bf16_t* wk1  = (bf16_t*)alloc(W1 * 2);
    bf16_t* wv1  = (bf16_t*)alloc(W1 * 2);
    bf16_t* wo1  = (bf16_t*)alloc(W1 * 2);
    bf16_t* wq2  = (bf16_t*)alloc(W1 * 2);
    bf16_t* wk2  = (bf16_t*)alloc(W1 * 2);
    bf16_t* wv2  = (bf16_t*)alloc(W1 * 2);
    bf16_t* wo2  = (bf16_t*)alloc(W1 * 2);
    bf16_t* wm1  = (bf16_t*)alloc(WM * 2);
    bf16_t* wm2  = (bf16_t*)alloc(WM * 2);
    bf16_t* embB = (bf16_t*)alloc((size_t)MTOK * D_MODEL * 2);
    float*  x_f  = (float*) alloc((size_t)MTOK * D_MODEL * 4);
    bf16_t* x_b  = (bf16_t*)alloc((size_t)MTOK * D_MODEL * 2);
    bf16_t* Qp   = (bf16_t*)alloc((size_t)MTOK * D_MODEL * 2);
    bf16_t* Kp   = (bf16_t*)alloc((size_t)MTOK * D_MODEL * 2);
    bf16_t* VpT  = (bf16_t*)alloc((size_t)MTOK * D_MODEL * 2);
    bf16_t* att1 = (bf16_t*)alloc((size_t)MTOK * D_MODEL * 2);
    float*  h_f  = (float*) alloc((size_t)MTOK * D_MODEL * 4);
    bf16_t* h2_b = (bf16_t*)alloc((size_t)MTOK * D_MODEL * 2);
    bf16_t* Qc   = (bf16_t*)alloc((size_t)MTOK * D_MODEL * 2);
    bf16_t* Kc   = (bf16_t*)alloc((size_t)MTOK * D_MODEL * 2);
    bf16_t* VcT  = (bf16_t*)alloc((size_t)MTOK * D_MODEL * 2);
    bf16_t* crs  = (bf16_t*)alloc((size_t)MTOK * D_MODEL * 2);
    float*  r_f  = (float*) alloc((size_t)MTOK * D_MODEL * 4);
    bf16_t* z_b  = (bf16_t*)alloc((size_t)MTOK * D_MODEL * 2);
    bf16_t* gact = (bf16_t*)alloc((size_t)MTOK * 4 * D_MODEL * 2);

    auto cvt = [&](const float* s, bf16_t* d, size_t n) {
        cvt_f32_bf16<<<(unsigned)((n + 255) / 256), 256, 0, stream>>>(s, d, (int)n);
    };
    cvt(q1_w,  wq1, W1);  cvt(k1_w, wk1, W1);  cvt(v1_w, wv1, W1);  cvt(out1_w, wo1, W1);
    cvt(q2_w,  wq2, W1);  cvt(k2_w, wk2, W1);  cvt(v2_w, wv2, W1);  cvt(out2_w, wo2, W1);
    cvt(mlp_w1, wm1, WM); cvt(mlp_w2, wm2, WM);
    cvt(input_embedding, embB, (size_t)MTOK * D_MODEL);

    auto gemm = [&](const bf16_t* A, const bf16_t* W, const float* bias, const float* res,
                    float* oF, bf16_t* oB, int M, int N, int K, int gelu, int trans) {
        dim3 grid(N / 128, M / 128);
        gemm_bf16_k<<<grid, 256, 0, stream>>>(A, W, bias, res, oF, oB, M, N, K, gelu, trans);
    };

    // x = LN1(tok_emb[target] + pos_emb)
    embed_ln_k<<<MTOK, 256, 0, stream>>>(target_ids, tok_emb, pos_emb, ln1_g, ln1_b, x_f, x_b);
    // self-attention
    gemm(x_b, wq1, q1_b, nullptr, nullptr, Qp,  MTOK, D_MODEL, D_MODEL, 0, 0);
    gemm(x_b, wk1, k1_b, nullptr, nullptr, Kp,  MTOK, D_MODEL, D_MODEL, 0, 0);
    gemm(x_b, wv1, v1_b, nullptr, nullptr, VpT, MTOK, D_MODEL, D_MODEL, 0, 1);
    flash_attn_k<<<512, 256, 0, stream>>>(Qp, Kp, VpT, target_ids, att1, 1);
    gemm(att1, wo1, out1_b, x_f, h_f, nullptr, MTOK, D_MODEL, D_MODEL, 0, 0);  // h = x + attn
    // cross-attention
    layernorm_k<<<MTOK, 256, 0, stream>>>(h_f, ln2_g, ln2_b, h2_b);
    gemm(h2_b, wq2, q2_b, nullptr, nullptr, Qc,  MTOK, D_MODEL, D_MODEL, 0, 0);
    gemm(embB, wk2, k2_b, nullptr, nullptr, Kc,  MTOK, D_MODEL, D_MODEL, 0, 0);
    gemm(embB, wv2, v2_b, nullptr, nullptr, VcT, MTOK, D_MODEL, D_MODEL, 0, 1);
    flash_attn_k<<<512, 256, 0, stream>>>(Qc, Kc, VcT, input_ids, crs, 0);
    gemm(crs, wo2, out2_b, h_f, r_f, nullptr, MTOK, D_MODEL, D_MODEL, 0, 0);   // r = cross + h
    // MLP
    layernorm_k<<<MTOK, 256, 0, stream>>>(r_f, ln3_g, ln3_b, z_b);
    gemm(z_b,  wm1, mlp_b1, nullptr, nullptr, gact, MTOK, 4 * D_MODEL, D_MODEL, 1, 0);
    gemm(gact, wm2, mlp_b2, r_f, (float*)d_out, nullptr, MTOK, D_MODEL, 4 * D_MODEL, 0, 0);
}
